// BatchGRU_51823075393626
// MI455X (gfx1250) — matrix operand order repaired and checked by reading, not compiled, not run on hardware
//
#include <hip/hip_runtime.h>
#include <hip/hip_bf16.h>

typedef unsigned short u16;
typedef __attribute__((ext_vector_type(16))) __bf16 v16bf;
typedef __attribute__((ext_vector_type(8)))  float  v8f;

constexpr int kH   = 512;   // hidden
constexpr int kB   = 512;   // batch (graphs)
constexpr int kT   = 128;   // padded length
constexpr int kK2  = 1024;  // concat K = [x | h]
constexpr int kWdir = 1572864; // ushorts of packed weights per direction
constexpr int LDA = 72;     // LDS pitch (ushorts); 144B rows: 16B aligned, bank-spread
constexpr int LDB = 72;

union Frag { uint4 q[2]; v16bf v; };

__device__ inline u16 f2bf(float f) {
  union { float f; unsigned u; } v; v.f = f;
  unsigned r = v.u + 0x7FFFu + ((v.u >> 16) & 1u);  // round-to-nearest-even
  return (u16)(r >> 16);
}
__device__ inline v8f wmma_bf16(const Frag& a, const Frag& b, v8f c) {
  // D = A(16x32 bf16) * B(32x16 bf16) + C(16x16 f32)
  return __builtin_amdgcn_wmma_f32_16x16x32_bf16(false, a.v, false, b.v,
                                                 (short)0, c, false, false);
}

// --- CDNA5 async global->LDS copy path (ASYNCcnt) ---------------------------
__device__ inline unsigned lds_off(const void* p) {
  // generic LDS address: low 32 bits are the LDS byte offset
  return (unsigned)(unsigned long long)p;
}
__device__ inline void async_copy_b128(unsigned ldsByteOff, const void* gaddr) {
  asm volatile("global_load_async_to_lds_b128 %0, %1, off"
               :: "v"(ldsByteOff), "v"(gaddr)
               : "memory");
}
__device__ inline void wait_async0() {
  asm volatile("s_wait_asynccnt 0x0" ::: "memory");
}

// ---------------------------------------------------------------------------
// Prep 1: starts[] prefix sum + fused gate biases  [dir][4][512]
//   g0: b_ih_r + b_hh_r, g1: b_ih_z + b_hh_z, g2: b_ih_n, g3: b_hh_n
// ---------------------------------------------------------------------------
__global__ __launch_bounds__(512) void prep_bias_starts_kernel(
    const int* __restrict__ lengths,
    const float* __restrict__ bihf, const float* __restrict__ bhhf,
    const float* __restrict__ bihb, const float* __restrict__ bhhb,
    int* __restrict__ starts, float* __restrict__ biasAll)
{
  const int tid = threadIdx.x;
  if (tid == 0) {
    int acc = 0;
    for (int i = 0; i < kB; ++i) { starts[i] = acc; acc += lengths[i]; }
  }
  if (tid < kH) {
    for (int d = 0; d < 2; ++d) {
      const float* bih = d ? bihb : bihf;
      const float* bhh = d ? bhhb : bhhf;
      float* bd = biasAll + d * 2048;
      bd[tid]        = bih[tid]        + bhh[tid];
      bd[512 + tid]  = bih[512 + tid]  + bhh[512 + tid];
      bd[1024 + tid] = bih[1024 + tid];
      bd[1536 + tid] = bhh[1024 + tid];
    }
  }
}

// ---------------------------------------------------------------------------
// Prep 2: repack weights to bf16, [n][k] with k contiguous.
//   Wr  [512][1024] = [w_ih_r | w_hh_r]   Wz likewise
//   Wnx [512][512]  = w_ih_n              Wnh [512][512] = w_hh_n
// ---------------------------------------------------------------------------
__global__ __launch_bounds__(256) void prep_weights_kernel(
    const float* __restrict__ wihf, const float* __restrict__ whhf,
    const float* __restrict__ wihb, const float* __restrict__ whhb,
    u16* __restrict__ Wall)
{
  const int n = blockIdx.x;           // output column 0..511
  const int d = blockIdx.y;           // direction
  const float* wih = d ? wihb : wihf;
  const float* whh = d ? whhb : whhf;
  u16* Wd = Wall + (size_t)d * kWdir;
  for (int k = threadIdx.x; k < kK2; k += 256) {
    const float vr = (k < kH) ? wih[(size_t)n * kH + k]
                              : whh[(size_t)n * kH + (k - kH)];
    const float vz = (k < kH) ? wih[(size_t)(kH + n) * kH + k]
                              : whh[(size_t)(kH + n) * kH + (k - kH)];
    Wd[(size_t)n * kK2 + k]          = f2bf(vr);
    Wd[524288 + (size_t)n * kK2 + k] = f2bf(vz);
  }
  for (int k = threadIdx.x; k < kH; k += 256) {
    Wd[1048576 + (size_t)n * kH + k] = f2bf(wih[(size_t)(2 * kH + n) * kH + k]);
    Wd[1310720 + (size_t)n * kH + k] = f2bf(whh[(size_t)(2 * kH + n) * kH + k]);
  }
}

// ---------------------------------------------------------------------------
// Prep 3: per-graph segment-max h0 (-> both dir hidden buffers, bf16+f32),
// padded relu(h+bias) input in bf16 [T][B][H], zero the sum accumulators.
// ---------------------------------------------------------------------------
__global__ __launch_bounds__(256) void prep_h0_pad_kernel(
    const float* __restrict__ h, const int* __restrict__ lengths,
    const int* __restrict__ starts, const float* __restrict__ bias0,
    u16* __restrict__ hbf, float* __restrict__ hfull,
    u16* __restrict__ xpad, float* __restrict__ sums)
{
  const int b = blockIdx.x;
  const int start = starts[b];
  const int len = lengths[b];
  const size_t HS = (size_t)kB * kH;      // slot stride
  for (int c = threadIdx.x; c < kH; c += 256) {
    float m = -3.4e38f;
    for (int i = 0; i < len; ++i)
      m = fmaxf(m, h[(size_t)(start + i) * kH + c]);
    const u16 mb = f2bf(m);
    hbf[(size_t)b * kH + c]            = mb;  // fwd ping-0
    hbf[2 * HS + (size_t)b * kH + c]   = mb;  // bwd ping-0
    hfull[(size_t)b * kH + c]          = m;
    hfull[2 * HS + (size_t)b * kH + c] = m;
  }
  for (int tt = 0; tt < kT; ++tt) {
    for (int c = threadIdx.x; c < kH; c += 256) {
      float v = 0.f;
      if (tt < len) v = fmaxf(h[(size_t)(start + tt) * kH + c] + bias0[c], 0.f);
      xpad[(size_t)tt * HS + (size_t)b * kH + c] = f2bf(v);
    }
  }
  for (int c = threadIdx.x; c < 2 * kH; c += 256)
    sums[(size_t)b * (2 * kH) + c] = 0.f;
}

// ---------------------------------------------------------------------------
// GRU step: one time index per direction. Grid (16,8,2): x=col tile (32),
// y=row tile (64), z=direction. 8 wave32s, each owns a 16x16 h tile with four
// WMMA f32 accumulators (r, z, n_x, n_h). K loop: 0..511 from x_t, 512..1023
// from h_prev; r/z accumulate over all K, n_x/n_h over their half only.
// LDS double-buffered; staging via global_load_async_to_lds_b128 (ASYNCcnt)
// overlapped with WMMA of the previous chunk.
// ---------------------------------------------------------------------------
__global__ __launch_bounds__(256) void gru_step_kernel(
    const u16* __restrict__ xpad, const u16* __restrict__ Wall,
    const float* __restrict__ biasAll,
    const u16* __restrict__ hfr, u16* __restrict__ hfw,
    const u16* __restrict__ hbr, u16* __restrict__ hbw,
    const float* __restrict__ Ffr, float* __restrict__ Ffw,
    const float* __restrict__ Fbr, float* __restrict__ Fbw,
    const int* __restrict__ lengths, float* __restrict__ sums, int step)
{
  const int dir = blockIdx.z;
  const int t = dir ? (kT - 1 - step) : step;
  const u16* Wd  = Wall + (size_t)dir * kWdir;
  const u16* Wr  = Wd;
  const u16* Wz  = Wd + 524288;
  const u16* Wnx = Wd + 1048576;
  const u16* Wnh = Wd + 1310720;
  const float* biasD = biasAll + dir * 2048;
  const u16* hread   = dir ? hbr : hfr;
  u16* hwrite        = dir ? hbw : hfw;
  const float* fread = dir ? Fbr : Ffr;
  float* fwrite      = dir ? Fbw : Ffw;

  __shared__ __align__(16) u16 ldsA [2][64 * LDA];
  __shared__ __align__(16) u16 ldsBr[2][32 * LDB];
  __shared__ __align__(16) u16 ldsBz[2][32 * LDB];
  __shared__ __align__(16) u16 ldsBn[2][32 * LDB];

  const int tid  = threadIdx.x;
  const int lane = tid & 31;
  const int wave = tid >> 5;
  const int wrow = wave & 3;            // 4 row groups of 16
  const int wcol = wave >> 2;           // 2 col groups of 16
  const int rowBase = blockIdx.y * 64;
  const int colBase = blockIdx.x * 32;
  const int hi = (lane >= 16) ? 1 : 0;

  v8f aR = {}; v8f aZ = {}; v8f aNX = {}; v8f aNH = {};

  const u16* xsrc = xpad + (size_t)t * ((size_t)kB * kH);

  // Per-thread async staging of one 64-wide K chunk into LDS buffer `buf`:
  // A tile 64x64 -> 2 x b128/thread, three B tiles 32x64 -> 3 x b128/thread.
  const int ar = tid >> 3;              // 0..31 (A row, +32 for second copy)
  const int ac = (tid & 7) << 3;        // 0,8,..,56 (ushort col)
  auto issue_chunk = [&](int buf, int kb) {
    const bool xh = (kb < kH);
    const u16* srcA = xh ? xsrc : hread;
    const u16* srcN = xh ? Wnx : Wnh;
    const int  kA   = kb & (kH - 1);
    async_copy_b128(lds_off(&ldsA[buf][ar * LDA + ac]),
                    srcA + (size_t)(rowBase + ar) * kH + kA + ac);
    async_copy_b128(lds_off(&ldsA[buf][(ar + 32) * LDA + ac]),
                    srcA + (size_t)(rowBase + ar + 32) * kH + kA + ac);
    async_copy_b128(lds_off(&ldsBr[buf][ar * LDB + ac]),
                    Wr + (size_t)(colBase + ar) * kK2 + kb + ac);
    async_copy_b128(lds_off(&ldsBz[buf][ar * LDB + ac]),
                    Wz + (size_t)(colBase + ar) * kK2 + kb + ac);
    async_copy_b128(lds_off(&ldsBn[buf][ar * LDB + ac]),
                    srcN + (size_t)(colBase + ar) * kH + kA + ac);
  };

  issue_chunk(0, 0);

  for (int kb = 0; kb < kK2; kb += 64) {
    const int cur = (kb >> 6) & 1;
    const bool xhalf = (kb < kH);

    wait_async0();        // our copies into buf[cur] are complete
    __syncthreads();      // everyone's copies done; everyone done reading buf[cur^1]

    if (kb + 64 < kK2) issue_chunk(cur ^ 1, kb + 64);  // overlap with compute

#pragma unroll
    for (int kk = 0; kk < 64; kk += 32) {
      Frag fa, fr, fz, fn;
      // A fragment: lane row = M, elems e -> K + e (+16 if lane>=16)
      // => two 16B ds loads, +0 and +16B
      const u16* pa = &ldsA[cur][(wrow * 16 + (lane & 15)) * LDA + kk + hi * 16];
      fa.q[0] = *(const uint4*)(pa);
      fa.q[1] = *(const uint4*)(pa + 8);
      // B fragment: lane col = N, elems e -> K + e (+16 if lane>=16)
      const int nb = (wcol * 16 + (lane & 15)) * LDB + kk + hi * 16;
      fr.q[0] = *(const uint4*)(&ldsBr[cur][nb]);
      fr.q[1] = *(const uint4*)(&ldsBr[cur][nb + 8]);
      fz.q[0] = *(const uint4*)(&ldsBz[cur][nb]);
      fz.q[1] = *(const uint4*)(&ldsBz[cur][nb + 8]);
      fn.q[0] = *(const uint4*)(&ldsBn[cur][nb]);
      fn.q[1] = *(const uint4*)(&ldsBn[cur][nb + 8]);
      aR = wmma_bf16(fa, fr, aR);
      aZ = wmma_bf16(fa, fz, aZ);
      if (xhalf) aNX = wmma_bf16(fa, fn, aNX);
      else       aNH = wmma_bf16(fa, fn, aNH);
    }
  }

  // Epilogue: gates + hidden update + masked mean accumulation.
  // D layout: elem e -> M = e + 8*(lane>=16), N = lane&15.
  const int n = colBase + wcol * 16 + (lane & 15);
  const float bR  = biasD[n];
  const float bZ  = biasD[512 + n];
  const float bNX = biasD[1024 + n];
  const float bNH = biasD[1536 + n];
  const int mBase = rowBase + wrow * 16 + hi * 8;
#pragma unroll
  for (int e = 0; e < 8; ++e) {
    const int m = mBase + e;
    const float r  = 1.f / (1.f + __expf(-(aR[e] + bR)));
    const float z  = 1.f / (1.f + __expf(-(aZ[e] + bZ)));
    const float nn = tanhf(aNX[e] + bNX + r * (aNH[e] + bNH));
    const float hp = fread[(size_t)m * kH + n];
    const float hn = (1.f - z) * nn + z * hp;
    fwrite[(size_t)m * kH + n] = hn;
    hwrite[(size_t)m * kH + n] = f2bf(hn);
    if (t < lengths[m])
      sums[(size_t)m * (2 * kH) + dir * kH + n] += hn;
  }
}

// ---------------------------------------------------------------------------
// Finalize: out[b, :] = sums[b, :] / len[b]
// ---------------------------------------------------------------------------
__global__ __launch_bounds__(256) void finalize_kernel(
    const float* __restrict__ sums, const int* __restrict__ lengths,
    float* __restrict__ out)
{
  const int b = blockIdx.x;
  const float inv = 1.f / (float)lengths[b];
  for (int c = threadIdx.x; c < 2 * kH; c += 256)
    out[(size_t)b * (2 * kH) + c] = sums[(size_t)b * (2 * kH) + c] * inv;
}

// ---------------------------------------------------------------------------
extern "C" void kernel_launch(void* const* d_in, const int* in_sizes, int n_in,
                              void* d_out, int out_size, void* d_ws, size_t ws_size,
                              hipStream_t stream) {
  (void)in_sizes; (void)n_in; (void)out_size; (void)ws_size;
  const float* h       = (const float*)d_in[0];
  const int*   lengths = (const int*)  d_in[1];
  const float* bias    = (const float*)d_in[2];
  const float* w_ih_f  = (const float*)d_in[3];
  const float* w_hh_f  = (const float*)d_in[4];
  const float* b_ih_f  = (const float*)d_in[5];
  const float* b_hh_f  = (const float*)d_in[6];
  const float* w_ih_b  = (const float*)d_in[7];
  const float* w_hh_b  = (const float*)d_in[8];
  const float* b_ih_b  = (const float*)d_in[9];
  const float* b_hh_b  = (const float*)d_in[10];

  char* ws = (char*)d_ws;
  size_t off = 0;
  auto take = [&](size_t bytes) -> char* {
    char* p = ws + off;
    off += (bytes + 255) & ~(size_t)255;
    return p;
  };
  int*   starts  = (int*)  take((size_t)kB * 4);
  float* biasAll = (float*)take((size_t)2 * 2048 * 4);
  float* sums    = (float*)take((size_t)kB * 2 * kH * 4);
  u16*   hbf     = (u16*)  take((size_t)4 * kB * kH * 2);   // 4 ping-pong slots
  float* hfull   = (float*)take((size_t)4 * kB * kH * 4);
  u16*   Wall    = (u16*)  take((size_t)2 * kWdir * 2);
  u16*   xpad    = (u16*)  take((size_t)kT * kB * kH * 2);
  // total workspace use: ~78 MB

  prep_bias_starts_kernel<<<1, 512, 0, stream>>>(
      lengths, b_ih_f, b_hh_f, b_ih_b, b_hh_b, starts, biasAll);
  prep_weights_kernel<<<dim3(512, 2), 256, 0, stream>>>(
      w_ih_f, w_hh_f, w_ih_b, w_hh_b, Wall);
  prep_h0_pad_kernel<<<512, 256, 0, stream>>>(
      h, lengths, starts, bias, hbf, hfull, xpad, sums);

  const size_t HS = (size_t)kB * kH;
  for (int s = 0; s < kT; ++s) {
    const int pr = s & 1, pw = pr ^ 1;
    gru_step_kernel<<<dim3(16, 8, 2), 256, 0, stream>>>(
        xpad, Wall, biasAll,
        hbf + (size_t)pr * HS,       hbf + (size_t)pw * HS,
        hbf + (size_t)(2 + pr) * HS, hbf + (size_t)(2 + pw) * HS,
        hfull + (size_t)pr * HS,       hfull + (size_t)pw * HS,
        hfull + (size_t)(2 + pr) * HS, hfull + (size_t)(2 + pw) * HS,
        lengths, sums, s);
  }
  finalize_kernel<<<512, 256, 0, stream>>>(sums, lengths, (float*)d_out);
}
